// SO2MessageBlock_38732015075525
// MI455X (gfx1250) — matrix operand order repaired and checked by reading, not compiled, not run on hardware
//
#include <hip/hip_runtime.h>
#include <hip/hip_bf16.h>
#include <math.h>

typedef __attribute__((ext_vector_type(16))) _Float16 v16h;
typedef __attribute__((ext_vector_type(8)))  _Float16 v8h;
typedef __attribute__((ext_vector_type(8)))  float    v8f;

#define DEV_INLINE __device__ __forceinline__

static constexpr int  B_  = 2;
static constexpr int  N_  = 128;
static constexpr int  D_  = 64;
static constexpr long E_  = (long)B_ * N_ * N_;   // 32768 edges
static constexpr int  BI_ = B_ * N_;              // 256 (b,i) pairs

// ---------------------------------------------------------------------------
// WMMA fragment helpers (gfx1250 wave32 layouts, CDNA5 ISA)
// A 16x32 f16 (MxK): lane -> M=lane&15, K = 16*half + s (same order as B)
// -> one contiguous 32-byte run per lane.
// ---------------------------------------------------------------------------
DEV_INLINE v16h load_a_frag(const _Float16* A, int ld, int r0, int k0) {
  const int lane = threadIdx.x & 31;
  const int half = lane >> 4;
  return *(const v16h*)(A + (r0 + (lane & 15)) * ld + k0 + (half << 4));
}

// B 32x16 f16 (KxN) from W^T stored [N][K] row-major: lane -> column n=lane&15,
// K = 16*half + 2v + t -> one contiguous 32-byte run per lane.
DEV_INLINE v16h load_b_frag(const _Float16* Wt, int ld, int k0, int n0) {
  const int lane = threadIdx.x & 31;
  const int half = lane >> 4;
  return *(const v16h*)(Wt + (n0 + (lane & 15)) * ld + k0 + (half << 4));
}

DEV_INLINE v8f wmma16(v16h a, v16h b, v8f c) {
  // args: (neg_a, A, neg_b, B, c_mod, C, reuse_a, reuse_b)
  return __builtin_amdgcn_wmma_f32_16x16x32_f16(false, a, false, b, (short)0, c,
                                                false, false);
}

DEV_INLINE void sh2v(float x, float y, float z, float* o) {
  const float s3 = 1.7320508075688772f;
  o[0] = s3 * x * z;
  o[1] = s3 * x * y;
  o[2] = y * y - 0.5f * (x * x + z * z);
  o[3] = s3 * y * z;
  o[4] = 0.5f * s3 * (z * z - x * x);
}

// ---------------------------------------------------------------------------
// Kernel A: edge MLP.  h = silu(rbf@w_r1+b1); w_all = (h@w_r2+b2)*mask
// One block per 128 edges, 8 waves.  WMMA 128x32x64 then 128x64x192.
// ---------------------------------------------------------------------------
__global__ __launch_bounds__(256) void so2_edge_mlp(
    const float* __restrict__ rbf, const float* __restrict__ mask,
    const float* __restrict__ w_r1, const float* __restrict__ b_r1,
    const float* __restrict__ w_r2, const float* __restrict__ b_r2,
    float* __restrict__ wall) {
  __shared__ _Float16 sA[128][32];    // rbf tile
  __shared__ _Float16 sW1[64][32];    // w_r1^T
  __shared__ _Float16 sH[128][64];    // hidden (post-silu)
  __shared__ _Float16 sW2[192][64];   // w_r2^T
  const int tid = threadIdx.x;
  const long e0 = (long)blockIdx.x * 128;

  for (int idx = tid; idx < 128 * 32; idx += 256)
    sA[idx >> 5][idx & 31] = (_Float16)rbf[e0 * 32 + idx];
  for (int idx = tid; idx < 32 * 64; idx += 256)
    sW1[idx & 63][idx >> 6] = (_Float16)w_r1[idx];          // [k*64+n] -> [n][k]
  for (int idx = tid; idx < 64 * 192; idx += 256)
    sW2[idx % 192][idx / 192] = (_Float16)w_r2[idx];        // [k*192+n] -> [n][k]
  __syncthreads();

  const int wave = tid >> 5, lane = tid & 31;
  const int half = lane >> 4, l16 = lane & 15;
  const int r0 = wave * 16;

  // GEMM1: M=128 N=64 K=32 (A-frag invariant across column tiles)
  {
    const v16h a = load_a_frag(&sA[0][0], 32, r0, 0);
    for (int nt = 0; nt < 4; ++nt) {
      const int n0 = nt * 16;
      v8f c = {};
      v16h b = load_b_frag(&sW1[0][0], 32, 0, n0);
      c = wmma16(a, b, c);
      const float bias = b_r1[n0 + l16];
#pragma unroll
      for (int v = 0; v < 8; ++v) {
        const int m = r0 + v + 8 * half;
        const float x = c[v] + bias;
        // silu with fast reciprocal (result goes to f16 anyway)
        sH[m][n0 + l16] = (_Float16)(x * __builtin_amdgcn_rcpf(1.0f + __expf(-x)));
      }
    }
  }
  __syncthreads();

  // GEMM2: M=128 N=192 K=64
  {
    v16h a0 = load_a_frag(&sH[0][0], 64, r0, 0);
    v16h a1 = load_a_frag(&sH[0][0], 64, r0, 32);
    for (int nt = 0; nt < 12; ++nt) {
      const int n0 = nt * 16;
      v8f c = {};
      c = wmma16(a0, load_b_frag(&sW2[0][0], 64, 0, n0), c);
      c = wmma16(a1, load_b_frag(&sW2[0][0], 64, 32, n0), c);
      const float bias = b_r2[n0 + l16];
#pragma unroll
      for (int v = 0; v < 8; ++v) {
        const int m = r0 + v + 8 * half;
        wall[(e0 + m) * 192 + n0 + l16] = (c[v] + bias) * mask[e0 + m];
      }
    }
  }
}

// ---------------------------------------------------------------------------
// Kernel B: per-edge geometry (R, Wigner D2) + build f16 GEMM inputs.
// One block per (b,i), one thread per edge j.
// ---------------------------------------------------------------------------
__global__ __launch_bounds__(128) void so2_geom(
    const float* __restrict__ node_vec, const float* __restrict__ node_tensor,
    const float* __restrict__ r_hat, const float* __restrict__ wall,
    float* __restrict__ Rout, float* __restrict__ D2out,
    _Float16* __restrict__ zin, _Float16* __restrict__ xyin,
    _Float16* __restrict__ m0in, _Float16* __restrict__ m1in,
    _Float16* __restrict__ m2in) {
  const int bi = blockIdx.x;
  const int j = threadIdx.x;
  const long e = (long)bi * 128 + j;

  __shared__ float nv[3][64];
  __shared__ float nt[5][64];
  for (int idx = j; idx < 192; idx += 128)
    nv[idx >> 6][idx & 63] = node_vec[(long)bi * 192 + idx];
  for (int idx = j; idx < 320; idx += 128)
    nt[idx >> 6][idx & 63] = node_tensor[(long)bi * 320 + idx];
  __syncthreads();

  // rotation_from_edge
  const float rx = r_hat[e * 3 + 0], ry = r_hat[e * 3 + 1], rz = r_hat[e * 3 + 2];
  const bool use_x = fabsf(rz) > 0.9f;
  const float refx = use_x ? 0.f : 1.f;
  const float refz = use_x ? 1.f : 0.f;
  const float proj = refx * rx + refz * rz;
  const float xr0 = refx - proj * rx;
  const float xr1 = -proj * ry;
  const float xr2 = refz - proj * rz;
  const float inr = 1.0f / (sqrtf(xr0 * xr0 + xr1 * xr1 + xr2 * xr2) + 1e-8f);
  const float x0 = xr0 * inr, x1 = xr1 * inr, x2 = xr2 * inr;
  const float y0 = ry * x2 - rz * x1;
  const float y1 = rz * x0 - rx * x2;
  const float y2 = rx * x1 - ry * x0;
  // R[p][q], columns = (x_norm, y_norm, r_hat)
  float R[3][3] = {{x0, y0, rx}, {x1, y1, ry}, {x2, y2, rz}};

  // Wigner D2: basis vectors (normalized rows of _V)
  const float s2 = 0.7071067811865476f;
  const float V[5][3] = {{1.f, 0.f, 0.f}, {0.f, s2, s2}, {0.f, 0.f, 1.f},
                         {s2, s2, 0.f},   {s2, 0.f, s2}};
  // A[c][k] = sh2(V[k])[c]; invert via Gauss-Jordan (same for all threads, cheap)
  float M[5][10];
  for (int k = 0; k < 5; ++k) {
    float o[5];
    sh2v(V[k][0], V[k][1], V[k][2], o);
    for (int c = 0; c < 5; ++c) M[c][k] = o[c];
  }
  for (int r = 0; r < 5; ++r)
    for (int c = 0; c < 5; ++c) M[r][5 + c] = (r == c) ? 1.f : 0.f;
  for (int col = 0; col < 5; ++col) {
    int piv = col;
    float best = fabsf(M[col][col]);
    for (int r = col + 1; r < 5; ++r) {
      const float av = fabsf(M[r][col]);
      if (av > best) { best = av; piv = r; }
    }
    if (piv != col)
      for (int c = 0; c < 10; ++c) {
        const float t = M[col][c]; M[col][c] = M[piv][c]; M[piv][c] = t;
      }
    const float inv = 1.f / M[col][col];
    for (int c = 0; c < 10; ++c) M[col][c] *= inv;
    for (int r = 0; r < 5; ++r)
      if (r != col) {
        const float f = M[r][col];
        for (int c = 0; c < 10; ++c) M[r][c] -= f * M[col][c];
      }
  }
  // Ap[k][c] = sh2(R @ V[k]);  D2[p][q] = sum_k Ap[k][p] * Ainv[k][q]
  float Ap[5][5];
  for (int k = 0; k < 5; ++k) {
    const float vx = R[0][0] * V[k][0] + R[0][1] * V[k][1] + R[0][2] * V[k][2];
    const float vy = R[1][0] * V[k][0] + R[1][1] * V[k][1] + R[1][2] * V[k][2];
    const float vz = R[2][0] * V[k][0] + R[2][1] * V[k][1] + R[2][2] * V[k][2];
    sh2v(vx, vy, vz, Ap[k]);
  }
  float D2[5][5];
  for (int p = 0; p < 5; ++p)
    for (int q = 0; q < 5; ++q) {
      float s = 0.f;
      for (int k = 0; k < 5; ++k) s += Ap[k][p] * M[k][5 + q];
      D2[p][q] = s;
    }

  for (int p = 0; p < 3; ++p)
    for (int q = 0; q < 3; ++q) Rout[e * 9 + p * 3 + q] = R[p][q];
  for (int p = 0; p < 5; ++p)
    for (int q = 0; q < 5; ++q) D2out[e * 25 + p * 5 + q] = D2[p][q];

  const float* wa = wall + e * 192;
  for (int d = 0; d < 64; ++d) {
    const float a0 = nv[0][d], a1 = nv[1][d], a2 = nv[2][d];
    const float vl0 = R[0][0] * a0 + R[0][1] * a1 + R[0][2] * a2;
    const float vl1 = R[1][0] * a0 + R[1][1] * a1 + R[1][2] * a2;
    const float vl2 = R[2][0] * a0 + R[2][1] * a1 + R[2][2] * a2;
    float tl[5];
#pragma unroll
    for (int p = 0; p < 5; ++p) {
      float s = 0.f;
#pragma unroll
      for (int q = 0; q < 5; ++q) s += D2[p][q] * nt[q][d];
      tl[p] = s;
    }
    const int dh = d >> 1;  // jnp.repeat(w,2)[c] == w[c>>1]
    const float w1full = wa[64 + d];
    const float w1a = wa[64 + dh], w1b = wa[64 + 32 + dh];
    const float w2full = wa[128 + d];
    const float w2a = wa[128 + dh], w2b = wa[128 + 32 + dh];
    zin[e * 64 + d]        = (_Float16)(vl2 * w1full);
    xyin[e * 128 + d]      = (_Float16)(vl0 * w1a);
    xyin[e * 128 + 64 + d] = (_Float16)(vl1 * w1b);
    m0in[e * 64 + d]       = (_Float16)(tl[2] * w2full);
    m1in[e * 128 + d]      = (_Float16)(tl[1] * w2a);
    m1in[e * 128 + 64 + d] = (_Float16)(tl[3] * w2b);
    m2in[e * 128 + d]      = (_Float16)(tl[0] * w2a);
    m2in[e * 128 + 64 + d] = (_Float16)(tl[4] * w2b);
  }
}

// ---------------------------------------------------------------------------
// Kernel C: fused [128xK]@[KxN] WMMA GEMM + contraction over edges with
// R^T / D2^T into per-(b,i) partial sums.  One block per (b,i).
// NCH = N/64 column halves (q channel constant per half), NROT = 3 or 5.
// Rotation tile is staged memory->LDS via the gfx1250 async path (ASYNCcnt).
// ---------------------------------------------------------------------------
template <int K, int NCH, int NROT>
__global__ __launch_bounds__(256) void so2_gemm_reduce(
    const _Float16* __restrict__ Ain, const float* __restrict__ W,
    const float* __restrict__ rot, float* __restrict__ partOut, int q0, int q1) {
  __shared__ __align__(16) float rt[128][NROT * NROT];
  __shared__ _Float16 Wt[64][K];
  __shared__ float part[8][2][NROT][64];

  const int bi = blockIdx.x;
  const long e0 = (long)bi * 128;
  const int tid = threadIdx.x;

  // prefetch edge activations for this block (global_prefetch_b8)
  __builtin_prefetch(Ain + (e0 + (tid & 127)) * K, 0, 3);

  // Async copy of the rotation tile into LDS: GLOBAL_LOAD_ASYNC_TO_LDS_B128,
  // no VGPR return data, tracked with ASYNCcnt (CDNA5 ISA).
  {
    constexpr int NF4 = (128 * NROT * NROT) / 4;  // float4 chunks (288 / 800)
    const float* gsrc = rot + e0 * NROT * NROT;
    for (int idx = tid; idx < NF4; idx += 256) {
      const unsigned lds_addr =
          (unsigned)(size_t)(&rt[0][0]) + (unsigned)idx * 16u;
      const float* g = gsrc + idx * 4;
      asm volatile("global_load_async_to_lds_b128 %0, %1, off"
                   :
                   : "v"(lds_addr), "v"(g)
                   : "memory");
    }
    asm volatile("s_wait_asynccnt 0x0" ::: "memory");
  }

  for (int idx = tid; idx < 8 * 2 * NROT * 64; idx += 256)
    (&part[0][0][0][0])[idx] = 0.f;
  __syncthreads();

  const int wave = tid >> 5, lane = tid & 31;
  const int half = lane >> 4, l16 = lane & 15;
  const int r0 = wave * 16;
  const int qs[2] = {q0, q1};
  const int N = NCH * 64;

  // A-fragments depend only on kk: load once, reuse for every column tile.
  v16h afr[K / 32];
#pragma unroll
  for (int kk = 0; kk < K / 32; ++kk)
    afr[kk] = load_a_frag(Ain + e0 * K, K, r0, kk * 32);

  for (int ch = 0; ch < NCH; ++ch) {
    for (int idx = tid; idx < 64 * K; idx += 256) {
      const int n = idx / K, k = idx % K;
      Wt[n][k] = (_Float16)W[k * N + ch * 64 + n];
    }
    __syncthreads();
    const int q = qs[ch];
    for (int nt = 0; nt < 4; ++nt) {
      const int n0 = nt * 16;
      v8f c = {};
#pragma unroll
      for (int kk = 0; kk < K / 32; ++kk)
        c = wmma16(afr[kk], load_b_frag(&Wt[0][0], K, kk * 32, n0), c);
      const int nf = n0 + l16;
#pragma unroll
      for (int v = 0; v < 8; ++v) {
        const int m = r0 + v + 8 * half;
        const float val = c[v];
#pragma unroll
        for (int p = 0; p < NROT; ++p)
          part[wave][half][p][nf] += rt[m][q * NROT + p] * val;  // R[q][p]
      }
    }
    __syncthreads();
  }
  for (int idx = tid; idx < NROT * 64; idx += 256) {
    const int p = idx >> 6, d = idx & 63;
    float s = 0.f;
#pragma unroll
    for (int w8 = 0; w8 < 8; ++w8)
      for (int h = 0; h < 2; ++h) s += part[w8][h][p][d];
    partOut[(long)bi * NROT * 64 + idx] = s;
  }
}

// ---------------------------------------------------------------------------
// Kernel D: scalar path + combine partials + 3 LayerNorms -> outputs.
// ---------------------------------------------------------------------------
__global__ __launch_bounds__(256) void so2_finalize(
    const float* __restrict__ node_scalar, const float* __restrict__ wall,
    const float* __restrict__ vpz, const float* __restrict__ vpxy,
    const float* __restrict__ tp0, const float* __restrict__ tp1,
    const float* __restrict__ tp2, const float* __restrict__ w_l0,
    const float* __restrict__ w_out, const float* __restrict__ b_out,
    const float* __restrict__ g_s, const float* __restrict__ beta_s,
    const float* __restrict__ g_v, const float* __restrict__ beta_v,
    const float* __restrict__ g_t, const float* __restrict__ beta_t,
    float* __restrict__ out) {
  const int bi = blockIdx.x;
  const int tid = threadIdx.x;
  const int d = tid & 63, grp = tid >> 6;
  const long e0 = (long)bi * 128;

  __shared__ float w0p[4][64];
  __shared__ float pre[9][64];   // row0=s, 1..3=v, 4..8=t
  __shared__ float msg_s[64];
  __shared__ float mu[9], rsig[9];

  float s = 0.f;
  for (int j = grp * 32; j < grp * 32 + 32; ++j) s += wall[(e0 + j) * 192 + d];
  w0p[grp][d] = s;

  if (grp < 3)
    pre[1 + grp][d] = vpz[(long)bi * 192 + grp * 64 + d] +
                      vpxy[(long)bi * 192 + grp * 64 + d];
  for (int p = grp; p < 5; p += 4)
    pre[4 + p][d] = tp0[(long)bi * 320 + p * 64 + d] +
                    tp1[(long)bi * 320 + p * 64 + d] +
                    tp2[(long)bi * 320 + p * 64 + d];
  __syncthreads();

  if (grp == 0) {
    const float w0s = w0p[0][d] + w0p[1][d] + w0p[2][d] + w0p[3][d];
    float sl = 0.f;
    for (int k = 0; k < 64; ++k) sl += node_scalar[(long)bi * 64 + k] * w_l0[k * 64 + d];
    msg_s[d] = sl * w0s;
  }
  __syncthreads();
  if (grp == 0) {
    float acc = b_out[d];
    for (int k = 0; k < 64; ++k) acc += msg_s[k] * w_out[k * 64 + d];
    pre[0][d] = acc;
  }
  __syncthreads();

  if (tid < 9) {
    float m = 0.f;
    for (int k = 0; k < 64; ++k) m += pre[tid][k];
    m *= (1.f / 64.f);
    float v = 0.f;
    for (int k = 0; k < 64; ++k) {
      const float df = pre[tid][k] - m;
      v += df * df;
    }
    v *= (1.f / 64.f);
    mu[tid] = m;
    rsig[tid] = rsqrtf(v + 1e-5f);
  }
  __syncthreads();

  for (int idx = tid; idx < 9 * 64; idx += 256) {
    const int r = idx >> 6, dd = idx & 63;
    const float x = (pre[r][dd] - mu[r]) * rsig[r];
    float g, bb;
    long off;
    if (r == 0) {
      g = g_s[dd]; bb = beta_s[dd];
      off = (long)bi * 64 + dd;
    } else if (r < 4) {
      g = g_v[dd]; bb = beta_v[dd];
      off = 16384 + ((long)bi * 3 + (r - 1)) * 64 + dd;
    } else {
      g = g_t[dd]; bb = beta_t[dd];
      off = 65536 + ((long)bi * 5 + (r - 4)) * 64 + dd;
    }
    out[off] = x * g + bb;
  }
}

// ---------------------------------------------------------------------------
extern "C" void kernel_launch(void* const* d_in, const int* in_sizes, int n_in,
                              void* d_out, int out_size, void* d_ws, size_t ws_size,
                              hipStream_t stream) {
  const float* node_scalar = (const float*)d_in[0];
  const float* node_vec    = (const float*)d_in[1];
  const float* node_tensor = (const float*)d_in[2];
  const float* rbf         = (const float*)d_in[3];
  const float* r_hat       = (const float*)d_in[4];
  const float* access_mask = (const float*)d_in[5];
  const float* w_r1   = (const float*)d_in[6];
  const float* b_r1   = (const float*)d_in[7];
  const float* w_r2   = (const float*)d_in[8];
  const float* b_r2   = (const float*)d_in[9];
  const float* w_l0   = (const float*)d_in[10];
  const float* w_l1m0 = (const float*)d_in[11];
  const float* w_l1m1 = (const float*)d_in[12];
  const float* w_l2m0 = (const float*)d_in[13];
  const float* w_l2m1 = (const float*)d_in[14];
  const float* w_l2m2 = (const float*)d_in[15];
  const float* w_out  = (const float*)d_in[16];
  const float* b_out  = (const float*)d_in[17];
  const float* g_s    = (const float*)d_in[18];
  const float* beta_s = (const float*)d_in[19];
  const float* g_v    = (const float*)d_in[20];
  const float* beta_v = (const float*)d_in[21];
  const float* g_t    = (const float*)d_in[22];
  const float* beta_t = (const float*)d_in[23];
  float* out = (float*)d_out;

  // Workspace layout (~65 MB)
  float* wall  = (float*)d_ws;            // E*192
  float* Rbuf  = wall + E_ * 192;         // E*9
  float* D2buf = Rbuf + E_ * 9;           // E*25
  float* vpz   = D2buf + E_ * 25;         // 256*3*64
  float* vpxy  = vpz + (long)BI_ * 192;
  float* tp0   = vpxy + (long)BI_ * 192;  // 256*5*64 each
  float* tp1   = tp0 + (long)BI_ * 320;
  float* tp2   = tp1 + (long)BI_ * 320;
  _Float16* zin  = (_Float16*)(tp2 + (long)BI_ * 320);  // E*64
  _Float16* xyin = zin + E_ * 64;                       // E*128
  _Float16* m0in = xyin + E_ * 128;                     // E*64
  _Float16* m1in = m0in + E_ * 64;                      // E*128
  _Float16* m2in = m1in + E_ * 128;                     // E*128
  (void)in_sizes; (void)n_in; (void)out_size; (void)ws_size;

  so2_edge_mlp<<<E_ / 128, 256, 0, stream>>>(rbf, access_mask, w_r1, b_r1, w_r2,
                                             b_r2, wall);
  so2_geom<<<BI_, 128, 0, stream>>>(node_vec, node_tensor, r_hat, wall, Rbuf,
                                    D2buf, zin, xyin, m0in, m1in, m2in);
  // l=1 channels (R reduction): z -> q=2 ; xy -> q=0,1
  so2_gemm_reduce<64, 1, 3><<<BI_, 256, 0, stream>>>(zin, w_l1m0, Rbuf, vpz, 2, 2);
  so2_gemm_reduce<128, 2, 3><<<BI_, 256, 0, stream>>>(xyin, w_l1m1, Rbuf, vpxy, 0, 1);
  // l=2 channels (D2 reduction): m0 -> q=2 ; m1 -> q=1,3 ; m2 -> q=0,4
  so2_gemm_reduce<64, 1, 5><<<BI_, 256, 0, stream>>>(m0in, w_l2m0, D2buf, tp0, 2, 2);
  so2_gemm_reduce<128, 2, 5><<<BI_, 256, 0, stream>>>(m1in, w_l2m1, D2buf, tp1, 1, 3);
  so2_gemm_reduce<128, 2, 5><<<BI_, 256, 0, stream>>>(m2in, w_l2m2, D2buf, tp2, 0, 4);
  so2_finalize<<<BI_, 256, 0, stream>>>(node_scalar, wall, vpz, vpxy, tp0, tp1,
                                        tp2, w_l0, w_out, b_out, g_s, beta_s,
                                        g_v, beta_v, g_t, beta_t, out);
}
